// graphNetwork_24292335026476
// MI455X (gfx1250) — matrix operand order repaired and checked by reading, not compiled, not run on hardware
//
#include <hip/hip_runtime.h>
#include <hip/hip_bf16.h>

#define NN      10000     // nodes
#define NE      160000    // edges
#define NNIN    40
#define NNOPENC 64
#define NEOPENC 32
#define NOPENC  128
#define NFEAT   640       // 5*NOPENC
#define NLAYERS 4
#define HSTEP   0.1f

typedef __attribute__((ext_vector_type(16))) _Float16 v16h;
typedef __attribute__((ext_vector_type(8)))  float    v8f;

// ---------------------------------------------------------------- utilities
__global__ void zero_k(float* p, int n) {
    int i = blockIdx.x * blockDim.x + threadIdx.x;
    if (i < n) p[i] = 0.0f;
}

__global__ void copy_k(const float* __restrict__ src, float* __restrict__ dst, int n) {
    int i = blockIdx.x * blockDim.x + threadIdx.x;
    if (i < n) dst[i] = src[i];
}

// --------------------------------------------------- weight pre-pack (f16, WMMA A-fragment layout)
// A-matrix 16x32 f16 layout: lane m (0..15) khalf=0: halves 0..15 = K0..15 ;
// lanes 16..31 (khalf=1): halves 0..15 = K16..31 (contiguous, matches B loads).
__device__ __forceinline__ int kmap(int i, int khalf) {
    return khalf * 16 + i;
}

__global__ void pack1_k(const float* __restrict__ KE1, _Float16* __restrict__ PA1) {
    int idx = blockIdx.x * blockDim.x + threadIdx.x;      // layers*8mt*20kt*32lane*16i
    if (idx >= NLAYERS * 8 * 20 * 32 * 16) return;
    int i    = idx & 15;
    int lane = (idx >> 4) & 31;
    int kt   = (idx >> 9) % 20;
    int mt   = (idx / (512 * 20)) % 8;
    int l    = idx / (512 * 20 * 8);
    int row  = mt * 16 + (lane & 15);
    int K    = kt * 32 + kmap(i, lane >> 4);
    PA1[idx] = (_Float16)KE1[((size_t)l * NOPENC + row) * NFEAT + K];
}

__global__ void pack2_k(const float* __restrict__ KE2, _Float16* __restrict__ PA2) {
    int idx = blockIdx.x * blockDim.x + threadIdx.x;      // layers*40mt*4kt*32lane*16i
    if (idx >= NLAYERS * 40 * 4 * 32 * 16) return;
    int i    = idx & 15;
    int lane = (idx >> 4) & 31;
    int kt   = (idx >> 9) & 3;
    int mt   = (idx / 2048) % 40;
    int l    = idx / 81920;
    int row  = mt * 16 + (lane & 15);
    int K    = kt * 32 + kmap(i, lane >> 4);
    PA2[idx] = (_Float16)KE2[((size_t)l * NFEAT + row) * NOPENC + K];
}

// --------------------------------------------------------------- node opener
// double_layer(xn, K1Nopen[64,40], K2Nopen[64,64]) ; one 64-thread block per node
__global__ __launch_bounds__(64) void node_open_k(
    const float* __restrict__ xin, const float* __restrict__ K1,
    const float* __restrict__ K2, float* __restrict__ XN)
{
    __shared__ float lin[NNIN];
    __shared__ float lmid[NNOPENC];
    int n = blockIdx.x, tid = threadIdx.x;
    if (tid < NNIN) lin[tid] = tanhf(xin[tid * NN + n]);
    __syncthreads();
    float y = 0.0f;
    for (int c = 0; c < NNIN; ++c) y += K1[tid * NNIN + c] * lin[c];
    lmid[tid] = y;
    __syncthreads();
    float s = 0.0f;
    for (int c = 0; c < NNOPENC; ++c) s += lmid[c];
    float mu = s * (1.0f / NNOPENC);
    float ss = 0.0f;
    for (int c = 0; c < NNOPENC; ++c) { float v = lmid[c] - mu; ss += v * v; }
    float iv = rsqrtf(ss + 1e-3f);
    __syncthreads();
    lmid[tid] = tanhf((y - mu) * iv);
    __syncthreads();
    float y2 = 0.0f;
    for (int c = 0; c < NNOPENC; ++c) y2 += K2[tid * NNOPENC + c] * lmid[c];
    XN[(size_t)tid * NN + n] = tanhf(y2);
}

// --------------------------------------------------------------- edge opener
__global__ void edge_open_k(const float* __restrict__ xe, const float* __restrict__ K1,
                            const float* __restrict__ K2, float* __restrict__ XEO)
{
    int e = blockIdx.x * blockDim.x + threadIdx.x;
    if (e >= NE) return;
    float x = tanhf(xe[e]);
    float y[NEOPENC], t[NEOPENC];
    float s = 0.0f;
#pragma unroll
    for (int c = 0; c < NEOPENC; ++c) { y[c] = K1[c] * x; s += y[c]; }
    float mu = s * (1.0f / NEOPENC);
    float ss = 0.0f;
#pragma unroll
    for (int c = 0; c < NEOPENC; ++c) { float v = y[c] - mu; ss += v * v; }
    float iv = rsqrtf(ss + 1e-3f);
#pragma unroll
    for (int c = 0; c < NEOPENC; ++c) t[c] = tanhf((y[c] - mu) * iv);
    for (int c = 0; c < NEOPENC; ++c) {
        float z = 0.0f;
#pragma unroll
        for (int k = 0; k < NEOPENC; ++k) z += K2[c * NEOPENC + k] * t[k];
        XEO[(size_t)c * NE + e] = tanhf(z);
    }
}

// ------------------------------------------- concat: edge_div (rows 64..95) & edge_ave scatters
__global__ void concat_scatter_k(const float* __restrict__ XEO,
                                 const int* __restrict__ EI, const int* __restrict__ EJ,
                                 float* __restrict__ XN, float* __restrict__ Z1, float* __restrict__ Z2)
{
    int e = blockIdx.x * blockDim.x + threadIdx.x;
    if (e >= NE) return;
    int ei = EI[e], ej = EJ[e];
    for (int c = 0; c < NEOPENC; ++c) {
        float v = XEO[(size_t)c * NE + e];
        atomicAdd(XN + (size_t)(NNOPENC + c) * NN + ei,  v);
        atomicAdd(XN + (size_t)(NNOPENC + c) * NN + ej, -v);
        atomicAdd(Z1 + (size_t)c * NN + ei, v);
        atomicAdd(Z2 + (size_t)c * NN + ej, v);
    }
}

__global__ void concat_max_k(const float* __restrict__ Z1, const float* __restrict__ Z2,
                             float* __restrict__ XN)
{
    int i = blockIdx.x * blockDim.x + threadIdx.x;      // over 32*NN
    if (i >= NEOPENC * NN) return;
    int c = i / NN, n = i % NN;
    XN[(size_t)(NNOPENC + NEOPENC + c) * NN + n] = fmaxf(Z1[(size_t)c * NN + n], Z2[(size_t)c * NN + n]);
}

// ---------------------------------------------------------------- coords x3 = KNclose @ xn
__global__ void coords_k(const float* __restrict__ XN, const float* __restrict__ KNc,
                         float* __restrict__ X3)
{
    int n = blockIdx.x * blockDim.x + threadIdx.x;
    if (n >= NN) return;
    for (int o = 0; o < 3; ++o) {
        float s = 0.0f;
        for (int c = 0; c < NOPENC; ++c) s += KNc[o * NOPENC + c] * XN[(size_t)c * NN + n];
        X3[(size_t)o * NN + n] = s;
    }
}

// ----------------------------------------------- edge lengths + block-reduced sum/sumsq
__global__ __launch_bounds__(256) void edgeL_k(const float* __restrict__ X3,
                                               const int* __restrict__ EI, const int* __restrict__ EJ,
                                               float* __restrict__ LT, float* __restrict__ SCAL)
{
    __shared__ float s0[256], s1[256];
    int e = blockIdx.x * 256 + threadIdx.x;
    float L = 0.0f;
    if (e < NE) {
        int ei = EI[e], ej = EJ[e];
        float d0 = X3[ei] - X3[ej];
        float d1 = X3[NN + ei] - X3[NN + ej];
        float d2 = X3[2 * NN + ei] - X3[2 * NN + ej];
        L = sqrtf(d0 * d0 + d1 * d1 + d2 * d2);
        LT[e] = L;
    }
    s0[threadIdx.x] = L; s1[threadIdx.x] = L * L;
    __syncthreads();
    for (int o = 128; o > 0; o >>= 1) {
        if (threadIdx.x < o) { s0[threadIdx.x] += s0[threadIdx.x + o]; s1[threadIdx.x] += s1[threadIdx.x + o]; }
        __syncthreads();
    }
    if (threadIdx.x == 0) { atomicAdd(SCAL, s0[0]); atomicAdd(SCAL + 1, s1[0]); }
}

// std over w = Kw @ L  (128*NE elements, ddof=1) ;  SCAL[2] = 1/(std+1e-4)
__global__ void std_k(float* SCAL, const float* __restrict__ KW) {
    double sumL = (double)SCAL[0], sumL2 = (double)SCAL[1];
    double sk = 0.0, sk2 = 0.0;
    for (int c = 0; c < NOPENC; ++c) { double k = (double)KW[c]; sk += k; sk2 += k * k; }
    double ntot = (double)NOPENC * (double)NE;
    double sw = sk * sumL, sw2 = sk2 * sumL2;
    double mean = sw / ntot;
    double ssd = sw2 - sw * mean;
    double var = ssd / (ntot - 1.0);
    if (var < 0.0) var = 0.0;
    SCAL[2] = (float)(1.0 / (sqrt(var) + 1e-4));
}

__global__ void edgew_k(float* __restrict__ LT, const float* __restrict__ SCAL) {
    int e = blockIdx.x * blockDim.x + threadIdx.x;
    if (e < NE) LT[e] = LT[e] * SCAL[2];          // t_e ; w[c,e] = exp(-(Kw[c]*t_e)^2)
}

// ===================================================================================
// fused edge layer: feature build -> WMMA GEMM1 (128x640) -> tv_norm -> WMMA GEMM2
// (640x128) -> weighted scatter (div + 4x max-scatter) ; last layer also emits xe_out
// one wave = 16 edges, 128-thread block = 64 edges
// ===================================================================================
__global__ __launch_bounds__(128) void edge_layer_k(
    const float* __restrict__ XN, const int* __restrict__ EI, const int* __restrict__ EJ,
    const float* __restrict__ TB, const float* __restrict__ KW,
    const _Float16* __restrict__ PA1, const _Float16* __restrict__ PA2,
    const float* __restrict__ KEc,
    float* __restrict__ DIV, float* __restrict__ ZB, float* __restrict__ xe_out,
    int layer, int last)
{
    extern __shared__ __align__(32) char smem[];
    const int wave = threadIdx.x >> 5, lane = threadIdx.x & 31;
    const int col = lane & 15, khalf = lane >> 4;
    char* wbase = smem + wave * 20480;
    _Float16* actA = (_Float16*)wbase;            // 16 cols x 640 K (col-major per edge)
    float*    nrm  = (float*)wbase;               // 16 cols x 128 (reuse after GEMM1)
    _Float16* act2 = (_Float16*)(wbase + 8192);   // 16 cols x 128
    float*    muv  = (float*)(wbase + 12288);
    float*    ivv  = (float*)(wbase + 12352);

    const int ebase  = blockIdx.x * 64 + wave * 16;
    const int myedge = ebase + col;
    const int eIl = EI[myedge], eJl = EJ[myedge];
    const float tE = TB[myedge];

    // ---- tanh(features) -> LDS (f16) ----
    for (int e16 = 0; e16 < 16; ++e16) {
        int e = ebase + e16;
        int ei = EI[e], ej = EJ[e];
        float t = TB[e];
        _Float16* dst = actA + e16 * NFEAT;
        for (int c = lane; c < NOPENC; c += 32) {
            float kt = KW[c] * t;
            float wce = __expf(-kt * kt);
            float xi = XN[(size_t)c * NN + ei], xj = XN[(size_t)c * NN + ej];
            float g = wce * (xi - xj);
            float a = wce * 0.5f * (xi + xj);
            dst[c]              = (_Float16)tanhf(g);
            dst[NOPENC + c]     = (_Float16)tanhf(a);
            dst[2 * NOPENC + c] = (_Float16)tanhf(g * a);
            dst[3 * NOPENC + c] = (_Float16)tanhf(g * g);
            dst[4 * NOPENC + c] = (_Float16)tanhf(a * a);
        }
    }

    // ---- GEMM1: out1[128][16] = KE1(128x640) * act(640x16) ----
    v8f acc[8] = {};
    const v16h* pa1 = (const v16h*)PA1 + (size_t)layer * 8 * 20 * 32;
    for (int kt = 0; kt < 20; ++kt) {
        v16h b = *(const v16h*)(actA + col * NFEAT + kt * 32 + khalf * 16);
#pragma unroll
        for (int mt = 0; mt < 8; ++mt) {
            v16h a = pa1[(mt * 20 + kt) * 32 + lane];
            acc[mt] = __builtin_amdgcn_wmma_f32_16x16x32_f16(
                false, a, false, b, (short)0, acc[mt], false, false);
        }
    }

    // ---- stage to LDS (col-major per edge), tv_norm over 128 ch, tanh, back to f16 ----
#pragma unroll
    for (int mt = 0; mt < 8; ++mt)
#pragma unroll
        for (int r = 0; r < 8; ++r)
            nrm[col * NOPENC + mt * 16 + khalf * 8 + r] = acc[mt][r];

    if (lane < 16) {
        float s = 0.0f;
        for (int m = 0; m < NOPENC; ++m) s += nrm[lane * NOPENC + m];
        float mu = s * (1.0f / NOPENC);
        float ss = 0.0f;
        for (int m = 0; m < NOPENC; ++m) { float v = nrm[lane * NOPENC + m] - mu; ss += v * v; }
        muv[lane] = mu;
        ivv[lane] = rsqrtf(ss + 1e-3f);
    }
    {
        float mu = muv[col], iv = ivv[col];
        for (int c = khalf * 64; c < khalf * 64 + 64; ++c)
            act2[col * NOPENC + c] = (_Float16)tanhf((nrm[col * NOPENC + c] - mu) * iv);
    }

    // ---- GEMM2: dxe[640][16] = KE2(640x128) * act2(128x16) ; tanh ; scatter ----
    v16h b2[4];
#pragma unroll
    for (int kt = 0; kt < 4; ++kt)
        b2[kt] = *(const v16h*)(act2 + col * NOPENC + kt * 32 + khalf * 16);

    const v16h* pa2 = (const v16h*)PA2 + (size_t)layer * 40 * 4 * 32;
    float xep = 0.0f;
    for (int mt = 0; mt < 40; ++mt) {
        v8f d = {};
#pragma unroll
        for (int kt = 0; kt < 4; ++kt) {
            v16h a = pa2[(mt * 4 + kt) * 32 + lane];
            d = __builtin_amdgcn_wmma_f32_16x16x32_f16(
                false, a, false, b2[kt], (short)0, d, false, false);
        }
        int mbase = mt * 16 + khalf * 8;
#pragma unroll
        for (int r = 0; r < 8; ++r) {
            int row = mbase + r;
            float val = tanhf(d[r]);
            int rr = row & (NOPENC - 1);
            float kt = KW[rr] * tE;
            float wE = __expf(-kt * kt);
            float wv = wE * val;
            if (row < NOPENC) {                       // edge_div chunk
                atomicAdd(DIV + (size_t)row * NN + eIl,  wv);
                atomicAdd(DIV + (size_t)row * NN + eJl, -wv);
            } else {                                  // edge_ave chunks k=1..4 (max of 2 scatters)
                int k = (row >> 7) - 1;
                atomicAdd(ZB + ((size_t)(2 * k)     * NOPENC + rr) * NN + eIl, wv);
                atomicAdd(ZB + ((size_t)(2 * k + 1) * NOPENC + rr) * NN + eJl, wv);
            }
            if (last) xep += KEc[row] * val;
        }
    }
    if (last) {
        xep += __shfl_xor(xep, 16, 32);
        if (lane < 16) xe_out[myedge] = xep;
    }
}

// -------------------------------------------- leapfrog state update
__global__ void xn_update_k(float* __restrict__ XN, float* __restrict__ XNOLD,
                            const float* __restrict__ DIV, const float* __restrict__ ZB)
{
    int i = blockIdx.x * blockDim.x + threadIdx.x;    // over 128*NN
    if (i >= NOPENC * NN) return;
    float agg = DIV[i];
    const size_t chunk = (size_t)NOPENC * NN;
#pragma unroll
    for (int k = 0; k < 4; ++k)
        agg += fmaxf(ZB[(size_t)(2 * k) * chunk + i], ZB[(size_t)(2 * k + 1) * chunk + i]);
    float xo = XNOLD[i], xc = XN[i];
    XNOLD[i] = xc;
    XN[i] = 2.0f * xc - xo - HSTEP * agg;
}

// -------------------------------------------- closing projection
__global__ void xn_close_k(const float* __restrict__ XN, const float* __restrict__ KNc,
                           float* __restrict__ out)
{
    int n = blockIdx.x * blockDim.x + threadIdx.x;
    if (n >= NN) return;
    for (int o = 0; o < 3; ++o) {
        float s = 0.0f;
        for (int c = 0; c < NOPENC; ++c) s += KNc[o * NOPENC + c] * XN[(size_t)c * NN + n];
        out[(size_t)o * NN + n] = s;
    }
}

// ===================================================================================
extern "C" void kernel_launch(void* const* d_in, const int* in_sizes, int n_in,
                              void* d_out, int out_size, void* d_ws, size_t ws_size,
                              hipStream_t stream) {
    (void)in_sizes; (void)n_in; (void)out_size; (void)ws_size;
    const float* xn_in = (const float*)d_in[0];
    const float* xe_in = (const float*)d_in[1];
    const int*   EI    = (const int*)d_in[2];
    const int*   EJ    = (const int*)d_in[3];
    const float* K1N   = (const float*)d_in[4];
    const float* K2N   = (const float*)d_in[5];
    const float* K1E   = (const float*)d_in[6];
    const float* K2E   = (const float*)d_in[7];
    const float* KE1   = (const float*)d_in[8];
    const float* KE2   = (const float*)d_in[9];
    const float* KNc   = (const float*)d_in[10];
    const float* KEc   = (const float*)d_in[11];
    const float* KW    = (const float*)d_in[12];
    float* outp = (float*)d_out;

    // workspace layout (floats)
    float* W     = (float*)d_ws;
    float* XN    = W;                                    // 128*NN
    float* XNOLD = XN + (size_t)NOPENC * NN;             // 128*NN
    float* XEO   = XNOLD + (size_t)NOPENC * NN;          // 32*NE
    float* X3    = XEO + (size_t)NEOPENC * NE;           // 3*NN
    float* LT    = X3 + (size_t)3 * NN;                  // NE
    float* SCAL  = LT + (size_t)NE;                      // 16
    float* DIV   = SCAL + 16;                            // 128*NN
    float* ZB    = DIV + (size_t)NOPENC * NN;            // 8 * 128*NN
    _Float16* PA1 = (_Float16*)(ZB + (size_t)8 * NOPENC * NN);          // 4*128*640 halves
    _Float16* PA2 = PA1 + (size_t)NLAYERS * NOPENC * NFEAT;             // 4*640*128 halves

    const int B = 256;
    const int nPack = NLAYERS * NOPENC * NFEAT;          // 327680

    // pack f16 WMMA-fragment weights
    pack1_k<<<(nPack + B - 1) / B, B, 0, stream>>>(KE1, PA1);
    pack2_k<<<(nPack + B - 1) / B, B, 0, stream>>>(KE2, PA2);

    // opening layers
    node_open_k<<<NN, 64, 0, stream>>>(xn_in, K1N, K2N, XN);
    edge_open_k<<<(NE + B - 1) / B, B, 0, stream>>>(xe_in, K1E, K2E, XEO);

    // concat: zero rows 64..127 of XN + all scatter buffers, scatter, max, copy xnold
    zero_k<<<((NNOPENC * NN) + B - 1) / B, B, 0, stream>>>(XN + (size_t)NNOPENC * NN, NNOPENC * NN);
    zero_k<<<((9 * NOPENC * NN) + B - 1) / B, B, 0, stream>>>(DIV, 9 * NOPENC * NN);
    concat_scatter_k<<<(NE + B - 1) / B, B, 0, stream>>>(XEO, EI, EJ, XN, ZB, ZB + (size_t)NOPENC * NN);
    concat_max_k<<<((NEOPENC * NN) + B - 1) / B, B, 0, stream>>>(ZB, ZB + (size_t)NOPENC * NN, XN);
    copy_k<<<((NOPENC * NN) + B - 1) / B, B, 0, stream>>>(XN, XNOLD, NOPENC * NN);

    // leapfrog layers
    for (int l = 0; l < NLAYERS; ++l) {
        coords_k<<<(NN + B - 1) / B, B, 0, stream>>>(XN, KNc, X3);
        zero_k<<<1, B, 0, stream>>>(SCAL, 4);
        edgeL_k<<<(NE + 255) / 256, 256, 0, stream>>>(X3, EI, EJ, LT, SCAL);
        std_k<<<1, 1, 0, stream>>>(SCAL, KW);
        edgew_k<<<(NE + B - 1) / B, B, 0, stream>>>(LT, SCAL);
        zero_k<<<((9 * NOPENC * NN) + B - 1) / B, B, 0, stream>>>(DIV, 9 * NOPENC * NN);
        edge_layer_k<<<NE / 64, 128, 4 * 20480, stream>>>(
            XN, EI, EJ, LT, KW, PA1, PA2, KEc, DIV, ZB, outp + 3 * NN, l, (l == NLAYERS - 1) ? 1 : 0);
        xn_update_k<<<((NOPENC * NN) + B - 1) / B, B, 0, stream>>>(XN, XNOLD, DIV, ZB);
    }

    // closing projections (xe_out written inside last edge_layer_k)
    xn_close_k<<<(NN + B - 1) / B, B, 0, stream>>>(XN, KNc, outp);
}
